// TransformerBlock_42743514530671
// MI455X (gfx1250) — matrix operand order repaired and checked by reading, not compiled, not run on hardware
//
#include <hip/hip_runtime.h>
#include <hip/hip_bf16.h>
#include <math.h>

// ---------------------------------------------------------------------------
// Types for WMMA fragments (wave32, v_wmma_f32_16x16x32_f16)
// ---------------------------------------------------------------------------
typedef __attribute__((ext_vector_type(16))) _Float16 v16h;
typedef __attribute__((ext_vector_type(8)))  _Float16 v8h;
typedef __attribute__((ext_vector_type(8)))  float    v8f;

union frag16 { v16h v; v8h h[2]; };
union f8     { v8f  v; float f[8]; };

static __device__ __forceinline__ v8f wmma_f16(const frag16& a, const frag16& b, v8f c) {
    return __builtin_amdgcn_wmma_f32_16x16x32_f16(false, a.v, false, b.v, (short)0, c, false, false);
}

// ---------------------------------------------------------------------------
// Problem constants
// ---------------------------------------------------------------------------
#define DMODEL 768
#define NHEAD  12
#define HD     64
#define FFDIM  3072
#define BATCH  4
#define SEQ    1024
#define TOKENS (BATCH * SEQ)   // 4096

// ---------------------------------------------------------------------------
// Weight fp32 -> fp16 convert + transpose:  wt[n*K + k] = (f16) w[k*N + n]
// ---------------------------------------------------------------------------
__global__ __launch_bounds__(256)
void cvtT_kernel(const float* __restrict__ w, _Float16* __restrict__ wt, int K, int N) {
    size_t i = (size_t)blockIdx.x * 256 + threadIdx.x;
    if (i >= (size_t)K * N) return;
    int n = (int)(i % N);
    int k = (int)(i / N);
    wt[(size_t)n * K + k] = (_Float16)w[i];
}

// ---------------------------------------------------------------------------
// LayerNorm over D=768, one row per 256-thread block, f16 output
// ---------------------------------------------------------------------------
__global__ __launch_bounds__(256)
void layernorm_kernel(const float* __restrict__ x, const float* __restrict__ g,
                      const float* __restrict__ be, _Float16* __restrict__ out) {
    __shared__ float red[256];
    const int row = blockIdx.x;
    const int t   = threadIdx.x;
    const float* xr = x + (size_t)row * DMODEL;
    float v0 = xr[t], v1 = xr[t + 256], v2 = xr[t + 512];
    red[t] = v0 + v1 + v2;
    __syncthreads();
    for (int o = 128; o > 0; o >>= 1) { if (t < o) red[t] += red[t + o]; __syncthreads(); }
    const float mu = red[0] * (1.0f / DMODEL);
    __syncthreads();
    float d0 = v0 - mu, d1 = v1 - mu, d2 = v2 - mu;
    red[t] = d0 * d0 + d1 * d1 + d2 * d2;
    __syncthreads();
    for (int o = 128; o > 0; o >>= 1) { if (t < o) red[t] += red[t + o]; __syncthreads(); }
    const float rinv = rsqrtf(red[0] * (1.0f / DMODEL) + 1e-5f);
    _Float16* orow = out + (size_t)row * DMODEL;
    orow[t]       = (_Float16)(d0 * rinv * g[t]       + be[t]);
    orow[t + 256] = (_Float16)(d1 * rinv * g[t + 256] + be[t + 256]);
    orow[t + 512] = (_Float16)(d2 * rinv * g[t + 512] + be[t + 512]);
}

// ---------------------------------------------------------------------------
// WMMA GEMM:  C[M,N] = A[M,K] * Bt[N,K]^T  (+ epilogue)
//   EPI 0: +bias                      -> f16
//   EPI 1: +bias, exact GELU          -> f16
//   EPI 2: +bias, +residual (f32)     -> f32
// Block: 256 threads = 8 waves (4m x 2n); tile 128x64, BK=32 staged in LDS.
// Each wave: 32x32 output = 4x v_wmma_f32_16x16x32_f16 per BK step.
// ---------------------------------------------------------------------------
#define BM 128
#define BN 64
#define BK 32

template<int EPI>
__global__ __launch_bounds__(256)
void gemm_kernel(const _Float16* __restrict__ A, const _Float16* __restrict__ Bt,
                 const float* __restrict__ bias, const float* __restrict__ res,
                 _Float16* __restrict__ out16, float* __restrict__ out32,
                 int M, int N, int K) {
    __shared__ __align__(16) _Float16 As[BM * BK];   // [row][k]
    __shared__ __align__(16) _Float16 Bs[BN * BK];   // [n][k]
    const int t    = threadIdx.x;
    const int bm   = blockIdx.x * BM;
    const int bn   = blockIdx.y * BN;
    const int wave = t >> 5, lane = t & 31;
    const int wm   = wave >> 1, wn = wave & 1;       // 4 x 2 wave grid
    const int hi   = lane >> 4, lm = lane & 15;

    f8 acc[2][2];
#pragma unroll
    for (int mt = 0; mt < 2; ++mt)
#pragma unroll
        for (int nt = 0; nt < 2; ++nt)
#pragma unroll
            for (int j = 0; j < 8; ++j) acc[mt][nt].f[j] = 0.0f;

    const int ar0 = t >> 2,          ap0 = t & 3;          // A chunk 0
    const int ar1 = (t + 256) >> 2,  ap1 = (t + 256) & 3;  // A chunk 1
    const int brn = t >> 2,          bpp = t & 3;          // B chunk

    for (int k0 = 0; k0 < K; k0 += BK) {
        __syncthreads();
        *(v8h*)&As[ar0 * BK + ap0 * 8] = *(const v8h*)&A[(size_t)(bm + ar0) * K + k0 + ap0 * 8];
        *(v8h*)&As[ar1 * BK + ap1 * 8] = *(const v8h*)&A[(size_t)(bm + ar1) * K + k0 + ap1 * 8];
        *(v8h*)&Bs[brn * BK + bpp * 8] = *(const v8h*)&Bt[(size_t)(bn + brn) * K + k0 + bpp * 8];
        __syncthreads();

        frag16 af[2], bf[2];
#pragma unroll
        for (int mt = 0; mt < 2; ++mt) {
            const _Float16* p = &As[(wm * 32 + mt * 16 + lm) * BK + hi * 8];
            af[mt].h[0] = *(const v8h*)p;
            af[mt].h[1] = *(const v8h*)(p + 16);
        }
#pragma unroll
        for (int nt = 0; nt < 2; ++nt) {
            const _Float16* p = &Bs[(wn * 32 + nt * 16 + lm) * BK + hi * 8];
            bf[nt].h[0] = *(const v8h*)p;
            bf[nt].h[1] = *(const v8h*)(p + 16);
        }
#pragma unroll
        for (int mt = 0; mt < 2; ++mt)
#pragma unroll
            for (int nt = 0; nt < 2; ++nt)
                acc[mt][nt].v = wmma_f16(af[mt], bf[nt], acc[mt][nt].v);
    }

    // epilogue
#pragma unroll
    for (int mt = 0; mt < 2; ++mt) {
#pragma unroll
        for (int nt = 0; nt < 2; ++nt) {
            const int col = bn + wn * 32 + nt * 16 + lm;
            const float bcol = bias[col];
#pragma unroll
            for (int j = 0; j < 8; ++j) {
                const int row = bm + wm * 32 + mt * 16 + j + hi * 8;
                float v = acc[mt][nt].f[j] + bcol;
                if (EPI == 1) v = 0.5f * v * (1.0f + erff(v * 0.70710678118f));
                if (EPI <= 1) {
                    out16[(size_t)row * N + col] = (_Float16)v;
                } else {
                    out32[(size_t)row * N + col] = v + res[(size_t)row * N + col];
                }
            }
        }
    }
}

// ---------------------------------------------------------------------------
// Split fused qkv [4096, 2304] f16 -> q[bh,n,d] (scaled 1/8), k[bh,n,d],
// vT[bh,d,n] (transposed so PV WMMA B-fragments are contiguous in key dim)
// ---------------------------------------------------------------------------
__global__ __launch_bounds__(256)
void split_kernel(const _Float16* __restrict__ qkv, _Float16* __restrict__ q,
                  _Float16* __restrict__ kk, _Float16* __restrict__ vT) {
    size_t i = (size_t)blockIdx.x * 256 + threadIdx.x;   // over 4096*2304
    const int c    = (int)(i % (3 * DMODEL));
    const size_t tok = i / (3 * DMODEL);
    const int which = c / DMODEL;
    const int h = (c % DMODEL) >> 6;
    const int d = c & 63;
    const int b = (int)(tok >> 10);
    const int n = (int)(tok & 1023);
    const size_t bh = (size_t)b * NHEAD + h;
    const _Float16 val = qkv[i];
    if (which == 0)      q[(bh * SEQ + n) * HD + d] = (_Float16)((float)val * 0.125f); // 1/sqrt(64)
    else if (which == 1) kk[(bh * SEQ + n) * HD + d] = val;
    else                 vT[(bh * HD + d) * SEQ + n] = val;
}

// ---------------------------------------------------------------------------
// Flash-style attention. Grid: 48 (b*h) * 16 query tiles; block: 4 waves.
// Each wave owns 16 queries; loops keys in steps of 32:
//   S = Q Kt  (2 C-frags, 4 WMMA), online softmax, P via LDS repack,
//   O += P V  (4 WMMA).
// ---------------------------------------------------------------------------
__global__ __launch_bounds__(128)
void attn_kernel(const _Float16* __restrict__ q, const _Float16* __restrict__ k,
                 const _Float16* __restrict__ vT, _Float16* __restrict__ o) {
    __shared__ __align__(16) _Float16 Pl[4][16 * 32];
    const int wave = threadIdx.x >> 5, lane = threadIdx.x & 31;
    const int hi = lane >> 4, lm = lane & 15;
    const int bh = blockIdx.x >> 4, qt = blockIdx.x & 15;
    const int b = bh / NHEAD, hh = bh % NHEAD;
    const int qbase = qt * 64 + wave * 16;
    const size_t kvbase = (size_t)bh * SEQ;

    // Q fragments (d = 0..31 and 32..63), kept resident
    frag16 aq[2];
    {
        const _Float16* qr = q + (kvbase + qbase + lm) * HD + hi * 8;
        aq[0].h[0] = *(const v8h*)qr;        aq[0].h[1] = *(const v8h*)(qr + 16);
        aq[1].h[0] = *(const v8h*)(qr + 32); aq[1].h[1] = *(const v8h*)(qr + 48);
    }

    f8 acc[4];
    float mrow[8], lrow[8];
#pragma unroll
    for (int nt = 0; nt < 4; ++nt)
#pragma unroll
        for (int j = 0; j < 8; ++j) acc[nt].f[j] = 0.0f;
#pragma unroll
    for (int j = 0; j < 8; ++j) { mrow[j] = -1e30f; lrow[j] = 0.0f; }

    for (int kt = 0; kt < SEQ; kt += 32) {
        // ---- scores for keys [kt, kt+16) and [kt+16, kt+32) ----
        f8 S0, S1;
        {
            frag16 b0, b1;
            const _Float16* kr = k + (kvbase + kt + lm) * HD + hi * 8;
            v8f z = {};
            b0.h[0] = *(const v8h*)kr;        b0.h[1] = *(const v8h*)(kr + 16);  // d 0..31
            b1.h[0] = *(const v8h*)(kr + 32); b1.h[1] = *(const v8h*)(kr + 48);  // d 32..63
            S0.v = wmma_f16(aq[0], b0, z);
            S0.v = wmma_f16(aq[1], b1, S0.v);
            const _Float16* kr2 = kr + 16 * HD;
            b0.h[0] = *(const v8h*)kr2;        b0.h[1] = *(const v8h*)(kr2 + 16);
            b1.h[0] = *(const v8h*)(kr2 + 32); b1.h[1] = *(const v8h*)(kr2 + 48);
            S1.v = wmma_f16(aq[0], b0, z);
            S1.v = wmma_f16(aq[1], b1, S1.v);
        }

        // ---- online softmax (row stats live per-lane, rows j + 8*hi) ----
#pragma unroll
        for (int j = 0; j < 8; ++j) {
            float mx = fmaxf(S0.f[j], S1.f[j]);
            mx = fmaxf(mx, __shfl_xor(mx, 1, 32));
            mx = fmaxf(mx, __shfl_xor(mx, 2, 32));
            mx = fmaxf(mx, __shfl_xor(mx, 4, 32));
            mx = fmaxf(mx, __shfl_xor(mx, 8, 32));
            const float nm = fmaxf(mrow[j], mx);
            const float sc = __expf(mrow[j] - nm);
            const float p0 = __expf(S0.f[j] - nm);
            const float p1 = __expf(S1.f[j] - nm);
            float rs = p0 + p1;
            rs += __shfl_xor(rs, 1, 32);
            rs += __shfl_xor(rs, 2, 32);
            rs += __shfl_xor(rs, 4, 32);
            rs += __shfl_xor(rs, 8, 32);
            lrow[j] = lrow[j] * sc + rs;
            mrow[j] = nm;
            acc[0].f[j] *= sc; acc[1].f[j] *= sc; acc[2].f[j] *= sc; acc[3].f[j] *= sc;
            Pl[wave][(j + hi * 8) * 32 + lm]      = (_Float16)p0;
            Pl[wave][(j + hi * 8) * 32 + lm + 16] = (_Float16)p1;
        }
        __syncthreads();

        // ---- O += P * V ----
        frag16 pa;
        {
            const _Float16* pr = &Pl[wave][lm * 32 + hi * 8];
            pa.h[0] = *(const v8h*)pr;
            pa.h[1] = *(const v8h*)(pr + 16);
        }
#pragma unroll
        for (int nt = 0; nt < 4; ++nt) {
            frag16 bv;
            const _Float16* vr = vT + ((size_t)bh * HD + nt * 16 + lm) * SEQ + kt + hi * 8;
            bv.h[0] = *(const v8h*)vr;
            bv.h[1] = *(const v8h*)(vr + 16);
            acc[nt].v = wmma_f16(pa, bv, acc[nt].v);
        }
        __syncthreads();
    }

    // ---- normalize and store o16 [token, 768] ----
#pragma unroll
    for (int j = 0; j < 8; ++j) {
        const float inv = 1.0f / lrow[j];
        const size_t tok = (size_t)b * SEQ + qbase + j + hi * 8;
#pragma unroll
        for (int nt = 0; nt < 4; ++nt)
            o[tok * DMODEL + hh * HD + nt * 16 + lm] = (_Float16)(acc[nt].f[j] * inv);
    }
}

// ---------------------------------------------------------------------------
// Launcher
// ---------------------------------------------------------------------------
extern "C" void kernel_launch(void* const* d_in, const int* in_sizes, int n_in,
                              void* d_out, int out_size, void* d_ws, size_t ws_size,
                              hipStream_t stream) {
    (void)in_sizes; (void)n_in; (void)out_size; (void)ws_size;
    const float* x      = (const float*)d_in[0];
    const float* w_qkv  = (const float*)d_in[1];
    const float* b_qkv  = (const float*)d_in[2];
    const float* w_proj = (const float*)d_in[3];
    const float* b_proj = (const float*)d_in[4];
    const float* w1     = (const float*)d_in[5];
    const float* b1     = (const float*)d_in[6];
    const float* w2     = (const float*)d_in[7];
    const float* b2     = (const float*)d_in[8];
    const float* g1     = (const float*)d_in[9];
    const float* be1    = (const float*)d_in[10];
    const float* g2     = (const float*)d_in[11];
    const float* be2    = (const float*)d_in[12];
    float* out = (float*)d_out;

    // workspace arena (all sizes already 256B multiples)
    char* ws = (char*)d_ws;
    size_t off = 0;
    _Float16* wqkvT  = (_Float16*)(ws + off); off += (size_t)DMODEL * 3 * DMODEL * 2;
    _Float16* wprojT = (_Float16*)(ws + off); off += (size_t)DMODEL * DMODEL * 2;
    _Float16* w1T    = (_Float16*)(ws + off); off += (size_t)DMODEL * FFDIM * 2;
    _Float16* w2T    = (_Float16*)(ws + off); off += (size_t)FFDIM * DMODEL * 2;
    _Float16* h16    = (_Float16*)(ws + off); off += (size_t)TOKENS * DMODEL * 2;   // reused for h2
    _Float16* rgnA   = (_Float16*)(ws + off); off += (size_t)TOKENS * FFDIM * 2;    // qkv16 then ff1
    _Float16* q16    = (_Float16*)(ws + off); off += (size_t)TOKENS * DMODEL * 2;
    _Float16* k16    = (_Float16*)(ws + off); off += (size_t)TOKENS * DMODEL * 2;
    _Float16* vT16   = (_Float16*)(ws + off); off += (size_t)TOKENS * DMODEL * 2;
    _Float16* o16    = (_Float16*)(ws + off); off += (size_t)TOKENS * DMODEL * 2;
    float*    x1     = (float*)   (ws + off); off += (size_t)TOKENS * DMODEL * 4;
    _Float16* qkv16  = rgnA;
    _Float16* ff1    = rgnA;

    // 1) weights -> f16, transposed [N,K]
    cvtT_kernel<<<(DMODEL * 3 * DMODEL) / 256, 256, 0, stream>>>(w_qkv,  wqkvT,  DMODEL, 3 * DMODEL);
    cvtT_kernel<<<(DMODEL * DMODEL)     / 256, 256, 0, stream>>>(w_proj, wprojT, DMODEL, DMODEL);
    cvtT_kernel<<<(DMODEL * FFDIM)      / 256, 256, 0, stream>>>(w1,     w1T,    DMODEL, FFDIM);
    cvtT_kernel<<<(FFDIM * DMODEL)      / 256, 256, 0, stream>>>(w2,     w2T,    FFDIM,  DMODEL);

    // 2) LN1
    layernorm_kernel<<<TOKENS, 256, 0, stream>>>(x, g1, be1, h16);

    // 3) qkv = h @ w_qkv + b
    gemm_kernel<0><<<dim3(TOKENS / BM, (3 * DMODEL) / BN), 256, 0, stream>>>(
        h16, wqkvT, b_qkv, nullptr, qkv16, nullptr, TOKENS, 3 * DMODEL, DMODEL);

    // 4) split -> q (scaled), k, vT
    split_kernel<<<((size_t)TOKENS * 3 * DMODEL) / 256, 256, 0, stream>>>(qkv16, q16, k16, vT16);

    // 5) attention
    attn_kernel<<<BATCH * NHEAD * 16, 128, 0, stream>>>(q16, k16, vT16, o16);

    // 6) x1 = x + o @ w_proj + b
    gemm_kernel<2><<<dim3(TOKENS / BM, DMODEL / BN), 256, 0, stream>>>(
        o16, wprojT, b_proj, x, nullptr, x1, TOKENS, DMODEL, DMODEL);

    // 7) LN2
    layernorm_kernel<<<TOKENS, 256, 0, stream>>>(x1, g2, be2, h16);

    // 8) ff1 = gelu(h2 @ w1 + b1)
    gemm_kernel<1><<<dim3(TOKENS / BM, FFDIM / BN), 256, 0, stream>>>(
        h16, w1T, b1, nullptr, ff1, nullptr, TOKENS, FFDIM, DMODEL);

    // 9) out = x1 + ff1 @ w2 + b2
    gemm_kernel<2><<<dim3(TOKENS / BM, DMODEL / BN), 256, 0, stream>>>(
        ff1, w2T, b2, x1, nullptr, out, TOKENS, DMODEL, FFDIM);
}
